// Alpha_Cos_GLM_43800076484946
// MI455X (gfx1250) — matrix operand order repaired and checked by reading, not compiled, not run on hardware
//
#include <hip/hip_runtime.h>
#include <hip/hip_bf16.h>
#include <math.h>

typedef __attribute__((ext_vector_type(16))) _Float16 v16h;
typedef __attribute__((ext_vector_type(8)))  float    v8f;

#define SUB   20
#define TNO   200
#define ENO   2000
#define INO   500
#define TD    20000
#define PITCH 32      // padded N stride for syn buffers
#define CBN   12
#define TB    128     // conv time-tile
#define CHUNK 200     // scan syn-input staging chunk
#define YBP   204     // ring-buffer row pitch (pad to dodge LDS bank conflicts)

// workspace layout (float offsets)
static constexpr size_t SYN_E_OFF = 0;          // 20000 x 32
static constexpr size_t SYN_I_OFF = 640000;     // 20000 x 32
static constexpr size_t SYNIN_OFF = 1280000;    // 20000 x 32
static constexpr size_t KE_OFF    = 1920000;    // 20 x 200
static constexpr size_t KI_OFF    = 1924000;    // 20 x 200
static constexpr size_t HK_OFF    = 1928000;    // 20 x 200 (unflipped hist kernel)
static constexpr size_t BPE_OFF   = 1932000;    // packed B (exc): 2*63*32*16 halfs = 32256 floats
static constexpr size_t BPI_OFF   = 1964256;    // packed B (inh): 2*16*32*16 halfs = 8192 floats

// output layout (float offsets)
static constexpr size_t FILT_OFF = 20000;
static constexpr size_t CSE_OFF  = 32000;
static constexpr size_t CSI_OFF  = 72000;

// ---------------------------------------------------------------------------
// 1) Filter precompute
// ---------------------------------------------------------------------------
__global__ void filters_kernel(const float* __restrict__ Wsyn,
                               const float* __restrict__ Tau,
                               const float* __restrict__ Delta,
                               const float* __restrict__ hw,
                               const float* __restrict__ hb,
                               float* __restrict__ ws,
                               float* __restrict__ out) {
  int s = blockIdx.x;     // 0..19
  int t = threadIdx.x;    // 0..199
  if (s >= SUB || t >= TNO) return;
  float ke = 0.f, ki = 0.f;
#pragma unroll
  for (int m = 0; m < 2; ++m) {
    float ue = fmaxf((float)t - Delta[s * 4 + m * 2 + 0], 0.f) * expf(-Tau[s * 4 + m * 2 + 0]);
    ke += ue * expf(-ue) * Wsyn[s * 4 + m * 2 + 0];
    float ui = fmaxf((float)t - Delta[s * 4 + m * 2 + 1], 0.f) * expf(-Tau[s * 4 + m * 2 + 1]);
    ki += ui * expf(-ui) * Wsyn[s * 4 + m * 2 + 1];
  }
  float hk = 0.f;
#pragma unroll
  for (int b = 0; b < CBN; ++b) hk += hw[s * CBN + b] * hb[b * TNO + t];

  ws[KE_OFF + s * TNO + t] = ke;
  ws[KI_OFF + s * TNO + t] = ki;
  ws[HK_OFF + s * TNO + t] = hk;
  out[FILT_OFF + (0  + s) * TNO + t]         = ke;
  out[FILT_OFF + (20 + s) * TNO + t]         = ki;
  out[FILT_OFF + (40 + s) * TNO + (TNO-1-t)] = hk;   // flipped
}

// ---------------------------------------------------------------------------
// 2a) Pre-pack B = C.T into fragment-major f16 so each lane's v16h B operand
//     is one contiguous 32-byte coalesced load. Tail K-tiles zero-padded.
//     Layout: [ntile(2)][ktile][lane(32)][h(16)] halfs.
// ---------------------------------------------------------------------------
template<int K>
__global__ __launch_bounds__(32) void pack_b_kernel(const float* __restrict__ C,
                                                    _Float16* __restrict__ Bp) {
  constexpr int NT = (K + 31) / 32;
  const int kt   = blockIdx.x;          // 0..NT-1
  const int lane = threadIdx.x;         // wave32
  const int n    = lane & 15;
  const int kb16 = (lane >> 4) * 16;
#pragma unroll
  for (int h = 0; h < 16; ++h) {
    int kk = kt * 32 + (h < 8 ? kb16 / 2 + h : 16 + kb16 / 2 + (h - 8));
    float x0 = 0.f, x1 = 0.f;
    if (kk < K) {
      x0 = C[(size_t)n * K + kk];
      if (n + 16 < SUB) x1 = C[(size_t)(n + 16) * K + kk];
    }
    Bp[((size_t)kt * 32 + lane) * 16 + h]        = (_Float16)x0;
    Bp[((size_t)(NT + kt) * 32 + lane) * 16 + h] = (_Float16)x1;
  }
}

// ---------------------------------------------------------------------------
// 2b) Spike projection GEMM via WMMA: OUT(TD x 32pad) = S(TD x K) @ C.T
//     K is a template constant -> full tiles have zero guards; A fragments are
//     4 aligned float4 loads per lane (two contiguous 8-float runs per the ISA
//     16-bit A layout); B fragments are single 32B loads from the packed
//     table. No LDS, no barriers in the hot loop.
// ---------------------------------------------------------------------------
template<int K>
__global__ __launch_bounds__(32) void spike_gemm_wmma(const float* __restrict__ S,
                                                      const _Float16* __restrict__ Bp,
                                                      float* __restrict__ OUT) {
  constexpr int NT     = (K + 31) / 32;
  constexpr int NFULL  = K / 32;
  constexpr int KFULL  = NFULL * 32;
  const int lane = threadIdx.x;
  const int row0 = blockIdx.x * 16;
  const int r    = lane & 15;
  const int hi   = lane >> 4;
  const int n    = r;
  const float* __restrict__ arow = S + (size_t)(row0 + r) * K;
  v8f c0 = {}; v8f c1 = {};

  for (int kt = 0; kt < NFULL; ++kt) {
    const int k0 = kt * 32;
    __builtin_prefetch(arow + k0 + 32, 0, 3);
    const float4 f0 = *(const float4*)(arow + k0 + 8 * hi);
    const float4 f1 = *(const float4*)(arow + k0 + 8 * hi + 4);
    const float4 f2 = *(const float4*)(arow + k0 + 16 + 8 * hi);
    const float4 f3 = *(const float4*)(arow + k0 + 16 + 8 * hi + 4);
    v16h a;
    a[0]  = (_Float16)f0.x; a[1]  = (_Float16)f0.y; a[2]  = (_Float16)f0.z; a[3]  = (_Float16)f0.w;
    a[4]  = (_Float16)f1.x; a[5]  = (_Float16)f1.y; a[6]  = (_Float16)f1.z; a[7]  = (_Float16)f1.w;
    a[8]  = (_Float16)f2.x; a[9]  = (_Float16)f2.y; a[10] = (_Float16)f2.z; a[11] = (_Float16)f2.w;
    a[12] = (_Float16)f3.x; a[13] = (_Float16)f3.y; a[14] = (_Float16)f3.z; a[15] = (_Float16)f3.w;

    v16h b0 = *(const v16h*)(Bp + ((size_t)kt * 32 + lane) * 16);
    v16h b1 = *(const v16h*)(Bp + ((size_t)(NT + kt) * 32 + lane) * 16);
    c0 = __builtin_amdgcn_wmma_f32_16x16x32_f16(false, a, false, b0, (short)0, c0, false, false);
    c1 = __builtin_amdgcn_wmma_f32_16x16x32_f16(false, a, false, b1, (short)0, c1, false, false);
  }
  if constexpr (K % 32 != 0) {
    // single tail tile; guards fold at compile time after unroll
    v16h a;
#pragma unroll
    for (int h = 0; h < 8; ++h) {
      int ka = KFULL + 8 * hi + h;
      int kb = KFULL + 16 + 8 * hi + h;
      a[h]     = (_Float16)((ka < K) ? arow[ka] : 0.f);
      a[8 + h] = (_Float16)((kb < K) ? arow[kb] : 0.f);
    }
    v16h b0 = *(const v16h*)(Bp + ((size_t)NFULL * 32 + lane) * 16);
    v16h b1 = *(const v16h*)(Bp + ((size_t)(NT + NFULL) * 32 + lane) * 16);
    c0 = __builtin_amdgcn_wmma_f32_16x16x32_f16(false, a, false, b0, (short)0, c0, false, false);
    c1 = __builtin_amdgcn_wmma_f32_16x16x32_f16(false, a, false, b1, (short)0, c1, false, false);
  }

  const int Mb = hi * 8;
#pragma unroll
  for (int rv = 0; rv < 8; ++rv) {
    OUT[(size_t)(row0 + Mb + rv) * PITCH + n]      = c0[rv];
    OUT[(size_t)(row0 + Mb + rv) * PITCH + 16 + n] = c1[rv];
  }
}

// ---------------------------------------------------------------------------
// 3) Depthwise causal conv, LDS-tiled; tile staging uses CDNA5 async
//    global->LDS copies (ASYNCcnt) instead of load+store through VGPRs.
// ---------------------------------------------------------------------------
__global__ void conv_kernel(const float* __restrict__ se_g,
                            const float* __restrict__ si_g,
                            const float* __restrict__ ke,
                            const float* __restrict__ ki,
                            float* __restrict__ synin) {
  __shared__ float se[(TB + TNO - 1) * SUB];
  __shared__ float si[(TB + TNO - 1) * SUB];
  const int t0 = blockIdx.x * TB;
  const int ROWS = TB + TNO - 1;

  for (int idx = threadIdx.x; idx < ROWS * SUB; idx += blockDim.x) {
    int i = idx / SUB, s = idx % SUB;
    int t = t0 - (TNO - 1) + i;
    if (t >= 0 && t < TD) {
      unsigned lds_e = (unsigned)(uintptr_t)&se[idx];
      unsigned lds_i = (unsigned)(uintptr_t)&si[idx];
      unsigned long long ga_e = (unsigned long long)(uintptr_t)(se_g + (size_t)t * PITCH + s);
      unsigned long long ga_i = (unsigned long long)(uintptr_t)(si_g + (size_t)t * PITCH + s);
      asm volatile("global_load_async_to_lds_b32 %0, %1, off"
                   :: "v"(lds_e), "v"(ga_e) : "memory");
      asm volatile("global_load_async_to_lds_b32 %0, %1, off"
                   :: "v"(lds_i), "v"(ga_i) : "memory");
    } else {
      se[idx] = 0.f;
      si[idx] = 0.f;
    }
  }
  asm volatile("s_wait_asynccnt 0x0" ::: "memory");
  __syncthreads();

  for (int o = threadIdx.x; o < TB * SUB; o += blockDim.x) {
    int tl = o / SUB, s = o % SUB;
    int t = t0 + tl;
    if (t >= TD) continue;
    const float* kes = ke + s * TNO;
    const float* kis = ki + s * TNO;
    float acc = 0.f;
    for (int tau = 0; tau < TNO; ++tau) {
      int i = tl + (TNO - 1) - tau;
      acc += kes[tau] * se[i * SUB + s] + kis[tau] * si[i * SUB + s];
    }
    synin[(size_t)t * PITCH + s] = acc;
  }
}

// ---------------------------------------------------------------------------
// 4) Sequential GLM scan — single wave, LDS ring buffer, chunked input staging
// ---------------------------------------------------------------------------
__global__ void scan_kernel(const float* __restrict__ synin,
                            const float* __restrict__ hk_g,
                            const float* __restrict__ Cden,
                            const float* __restrict__ Wsub,
                            const float* __restrict__ Theta,
                            const float* __restrict__ Vo,
                            float* __restrict__ out) {
  __shared__ float ybuf[SUB][YBP];
  __shared__ float hk[SUB][TNO];
  __shared__ float cw[SUB][SUB];
  __shared__ float sin_l[CHUNK][SUB];
  __shared__ float theta_l[SUB];
  const int lane = threadIdx.x;

  for (int idx = lane; idx < SUB * YBP; idx += 32) ((float*)ybuf)[idx] = 0.f;
  for (int idx = lane; idx < SUB * TNO; idx += 32) ((float*)hk)[idx] = hk_g[idx];
  for (int idx = lane; idx < SUB * SUB; idx += 32) {
    int s = idx / SUB, p = idx % SUB;
    cw[s][p] = Cden[s * SUB + p] * Wsub[p];
  }
  if (lane < SUB) theta_l[lane] = Theta[lane];
  const float w0 = Wsub[0];
  const float vo = Vo[0];
  __syncthreads();

  const int s = lane;
  for (int t = 0; t < TD; ++t) {
    const int slot = t % TNO;
    if ((t % CHUNK) == 0) {
      __syncthreads();
      for (int idx = lane; idx < CHUNK * SUB; idx += 32) {
        int j = idx / SUB, ss = idx % SUB;
        int tt = t + j;
        sin_l[j][ss] = (tt < TD) ? synin[(size_t)tt * PITCH + ss] : 0.f;
      }
      __syncthreads();
    }
    float y = 0.f;
    if (s < SUB) {
      float acc = 0.f;
      for (int j = slot - 1; j >= 0; --j)   acc += ybuf[s][j] * hk[s][slot - 1 - j];
      for (int j = TNO - 1; j >= slot; --j) acc += ybuf[s][j] * hk[s][slot + TNO - 1 - j];
      const int pslot = (t + TNO - 1) % TNO;
      float prop = 0.f;
#pragma unroll
      for (int p = 0; p < SUB; ++p) prop += ybuf[p][pslot] * cw[s][p];
      y = tanhf(sin_l[t % CHUNK][s] + prop + theta_l[s] + acc);
    }
    __syncthreads();
    if (s < SUB) ybuf[s][slot] = y;
    if (s == 0)  out[t] = w0 * y + vo;
    __syncthreads();
  }
}

// ---------------------------------------------------------------------------
extern "C" void kernel_launch(void* const* d_in, const int* in_sizes, int n_in,
                              void* d_out, int out_size, void* d_ws, size_t ws_size,
                              hipStream_t stream) {
  (void)in_sizes; (void)n_in; (void)out_size; (void)ws_size;
  const float* S_e     = (const float*)d_in[0];
  const float* S_i     = (const float*)d_in[1];
  const float* C_den   = (const float*)d_in[2];
  const float* C_syn_e = (const float*)d_in[3];
  const float* C_syn_i = (const float*)d_in[4];
  const float* W_syn   = (const float*)d_in[5];
  const float* Tau_syn = (const float*)d_in[6];
  const float* Delta   = (const float*)d_in[7];
  const float* W_sub   = (const float*)d_in[8];
  const float* V_o     = (const float*)d_in[9];
  const float* Theta   = (const float*)d_in[10];
  const float* hist_w  = (const float*)d_in[11];
  const float* hist_b  = (const float*)d_in[12];

  float* out = (float*)d_out;
  float* ws  = (float*)d_ws;
  float* syn_e  = ws + SYN_E_OFF;
  float* syn_i  = ws + SYN_I_OFF;
  float* syn_in = ws + SYNIN_OFF;
  _Float16* bp_e = (_Float16*)(ws + BPE_OFF);
  _Float16* bp_i = (_Float16*)(ws + BPI_OFF);

  filters_kernel<<<dim3(SUB), dim3(TNO), 0, stream>>>(W_syn, Tau_syn, Delta,
                                                      hist_w, hist_b, ws, out);
  pack_b_kernel<ENO><<<dim3((ENO + 31) / 32), dim3(32), 0, stream>>>(C_syn_e, bp_e);
  pack_b_kernel<INO><<<dim3((INO + 31) / 32), dim3(32), 0, stream>>>(C_syn_i, bp_i);
  spike_gemm_wmma<ENO><<<dim3(TD / 16), dim3(32), 0, stream>>>(S_e, bp_e, syn_e);
  spike_gemm_wmma<INO><<<dim3(TD / 16), dim3(32), 0, stream>>>(S_i, bp_i, syn_i);
  conv_kernel<<<dim3((TD + TB - 1) / TB), dim3(256), 0, stream>>>(
      syn_e, syn_i, ws + KE_OFF, ws + KI_OFF, syn_in);
  scan_kernel<<<dim3(1), dim3(32), 0, stream>>>(syn_in, ws + HK_OFF, C_den,
                                                W_sub, V_o, Theta, out);
  hipMemcpyAsync(out + CSE_OFF, C_syn_e, (size_t)SUB * ENO * sizeof(float),
                 hipMemcpyDeviceToDevice, stream);
  hipMemcpyAsync(out + CSI_OFF, C_syn_i, (size_t)SUB * INO * sizeof(float),
                 hipMemcpyDeviceToDevice, stream);
}
